// ProductionEdgeLevelGNN_807453851682
// MI455X (gfx1250) — matrix operand order repaired and checked by reading, not compiled, not run on hardware
//
#include <hip/hip_runtime.h>
#include <hip/hip_bf16.h>

#define NNODE 50000
#define NEDGE 800000
#define DIM 128

typedef __bf16 bf16_t;
typedef __attribute__((ext_vector_type(16))) __bf16 v16bf;
typedef __attribute__((ext_vector_type(8)))  __bf16 v8bf;
typedef __attribute__((ext_vector_type(8)))  float  v8f;

// ---------------- graph norm ----------------
__global__ void k_degree(const int* __restrict__ dst, float* __restrict__ deg) {
  int t = blockIdx.x * blockDim.x + threadIdx.x;
  if (t < NEDGE) atomicAdd(&deg[dst[t]], 1.0f);
}

__global__ void k_dinv(const float* __restrict__ deg, float* __restrict__ dinv) {
  int t = blockIdx.x * blockDim.x + threadIdx.x;
  if (t < NNODE) dinv[t] = rsqrtf(deg[t] + 1.0f);  // +1 self-loop
}

// ---------------- weight prep: W[K,N] f32 -> Bt[Npad,K] bf16 (transposed) ----------------
__global__ void k_transpose_bf16(const float* __restrict__ W, bf16_t* __restrict__ Bt,
                                 int K, int Nsrc, int Npad) {
  int t = blockIdx.x * blockDim.x + threadIdx.x;
  if (t >= Npad * K) return;
  int n = t / K, k = t - n * K;
  float v = (n < Nsrc) ? W[(long)k * Nsrc + n] : 0.0f;
  Bt[t] = (bf16_t)v;
}

__global__ void k_cvt_bf16(const float* __restrict__ in, bf16_t* __restrict__ out, int n) {
  int t = blockIdx.x * blockDim.x + threadIdx.x;
  if (t < n) out[t] = (bf16_t)in[t];
}

// ---------------- aggregation ----------------
__global__ void k_init_agg(const float* __restrict__ xw, const float* __restrict__ dinv,
                           const float* __restrict__ b, float* __restrict__ agg) {
  int t = blockIdx.x * blockDim.x + threadIdx.x;
  if (t >= NNODE * DIM) return;
  int n = t >> 7, f = t & 127;
  float di = dinv[n];
  agg[t] = xw[t] * di * di + b[f];  // self-loop contribution + bias
}

__global__ void k_scatter(const float* __restrict__ xw, const float* __restrict__ dinv,
                          const int* __restrict__ src, const int* __restrict__ dst,
                          float* __restrict__ agg) {
  long t = (long)blockIdx.x * blockDim.x + threadIdx.x;  // 32 lanes per edge
  int e = (int)(t >> 5), l = (int)(t & 31);
  int s = src[e], d = dst[e];
  float nrm = dinv[s] * dinv[d];
  const float4 v = *(const float4*)(xw + (long)s * DIM + l * 4);
  float* p = agg + (long)d * DIM + l * 4;
  atomicAdd(p + 0, v.x * nrm);
  atomicAdd(p + 1, v.y * nrm);
  atomicAdd(p + 2, v.z * nrm);
  atomicAdd(p + 3, v.w * nrm);
}

// ---------------- batchnorm (training-mode, biased var) ----------------
__global__ void k_bn_stats(const float* __restrict__ agg, float* __restrict__ mean,
                           float* __restrict__ rstd) {
  __shared__ float ss[256], sq[256];
  int f = blockIdx.x, t = threadIdx.x;
  float s = 0.f, q = 0.f;
  for (int n = t; n < NNODE; n += 256) {
    float v = agg[(long)n * DIM + f];
    s += v; q += v * v;
  }
  ss[t] = s; sq[t] = q;
  __syncthreads();
  for (int o = 128; o > 0; o >>= 1) {
    if (t < o) { ss[t] += ss[t + o]; sq[t] += sq[t + o]; }
    __syncthreads();
  }
  if (t == 0) {
    float m = ss[0] / (float)NNODE;
    float var = sq[0] / (float)NNODE - m * m;
    mean[f] = m;
    rstd[f] = rsqrtf(var + 1e-5f);
  }
}

__global__ void k_bn_apply(const float* __restrict__ agg, const float* __restrict__ mean,
                           const float* __restrict__ rstd, const float* __restrict__ g,
                           const float* __restrict__ beta, bf16_t* __restrict__ out) {
  int t = blockIdx.x * blockDim.x + threadIdx.x;
  if (t >= NNODE * DIM) return;
  int f = t & 127;
  float v = (agg[t] - mean[f]) * rstd[f] * g[f] + beta[f];
  out[t] = (bf16_t)fmaxf(v, 0.0f);  // ReLU fused, bf16 for next WMMA GEMM
}

// ---------------- WMMA bf16 GEMM with LDS-staged B ----------------
// C[M,Nc] = act(A[M,K] @ Bt^T + bias). 256-thread block = 8 waves.
// The whole Bt (Nc x K, bf16) is cooperatively staged into LDS once per block
// (rows padded by 16 bf16 = 8-bank rotation so the 16 columns read by lanes
// 0-15 land in distinct banks). Each wave then owns one 16-row tile and all
// Nc columns; the K-loop issues ds_load_b128 pairs (addresses independent of
// prior WMMAs -> schedulable early, ~LDS latency) instead of per-WMMA L2
// round trips. A is fetched from global (per-wave, L2-resident), amortized
// over NT WMMAs. GATHER mode builds A row r = concat(h[gsrc[r]], h[gdst[r]]).
template <int KT, int NT, bool GATHER, bool RELU, bool BF16OUT, int OUTCOLS>
__global__ void __launch_bounds__(256) k_gemm(
    const bf16_t* __restrict__ A, const bf16_t* __restrict__ Bt,
    const float* __restrict__ bias, float* __restrict__ Cf, bf16_t* __restrict__ Cb,
    const int* __restrict__ gsrc, const int* __restrict__ gdst, int Mtiles) {
  constexpr int K = KT * 32;
  constexpr int Nc = NT * 16;
  constexpr int LDK = K + 16;      // padded LDS row stride (bf16 elems)
  constexpr int KCH = K / 8;       // 16B chunks per row

  extern __shared__ __align__(16) bf16_t Bs[];  // Nc * LDK bf16

  // cooperative B fill (all threads; BEFORE any early exit — barrier safety)
  for (int c = threadIdx.x; c < Nc * KCH; c += 256) {
    int n = c / KCH, k8 = c - n * KCH;
    *(v8bf*)(Bs + n * LDK + k8 * 8) = *(const v8bf*)(Bt + (long)n * K + k8 * 8);
  }
  __syncthreads();

  const int lane = threadIdx.x & 31;
  const int wave = threadIdx.x >> 5;
  const int mt = blockIdx.x * 8 + wave;
  if (mt >= Mtiles) return;  // wave-uniform: EXEC stays all-ones for WMMA
  const int col16 = lane & 15;
  const int hsel = lane >> 4;
  const int rglob = mt * 16 + col16;

  const bf16_t* pS;
  const bf16_t* pD = nullptr;
  if constexpr (GATHER) {
    pS = A + (long)gsrc[rglob] * (K / 2);
    pD = A + (long)gdst[rglob] * (K / 2);
    __builtin_prefetch(pS, 0, 0);   // global_prefetch_b8
    __builtin_prefetch(pD, 0, 0);
  } else {
    pS = A + (long)rglob * K;
  }

  v8f acc[NT];
#pragma unroll
  for (int nt = 0; nt < NT; ++nt)
#pragma unroll
    for (int i = 0; i < 8; ++i) acc[nt][i] = 0.0f;

  for (int kt = 0; kt < KT; ++kt) {
    const int k0 = kt * 32 + 8 * hsel;
    const int k1 = k0 + 16;
    const v8bf* q0;
    const v8bf* q1;
    if constexpr (GATHER) {  // 8-elem chunks never straddle the src/dst halves
      q0 = (const v8bf*)((k0 < K / 2) ? (pS + k0) : (pD + (k0 - K / 2)));
      q1 = (const v8bf*)((k1 < K / 2) ? (pS + k1) : (pD + (k1 - K / 2)));
    } else {
      q0 = (const v8bf*)(pS + k0);
      q1 = (const v8bf*)(pS + k1);
    }
    v8bf alo = *q0, ahi = *q1;
    v16bf a;
#pragma unroll
    for (int i = 0; i < 8; ++i) { a[i] = alo[i]; a[8 + i] = ahi[i]; }
#pragma unroll
    for (int nt = 0; nt < NT; ++nt) {
      const bf16_t* pB = Bs + (nt * 16 + col16) * LDK + kt * 32 + 8 * hsel;
      v8bf blo = *(const v8bf*)pB;       // ds_load_b128
      v8bf bhi = *(const v8bf*)(pB + 16); // ds_load_b128
      v16bf b;
#pragma unroll
      for (int i = 0; i < 8; ++i) { b[i] = blo[i]; b[8 + i] = bhi[i]; }
      acc[nt] = __builtin_amdgcn_wmma_f32_16x16x32_bf16(
          false, a, false, b, (short)0, acc[nt], false, false);
    }
  }

#pragma unroll
  for (int nt = 0; nt < NT; ++nt) {
    const int ocol = nt * 16 + col16;
    float bv = 0.0f;
    if constexpr (OUTCOLS > 0) {
      if (ocol < OUTCOLS) bv = bias[ocol];  // padded columns have no bias
    } else {
      bv = bias ? bias[ocol] : 0.0f;
    }
#pragma unroll
    for (int r = 0; r < 8; ++r) {
      const int orow = mt * 16 + r + 8 * hsel;  // C/D layout: M = r + 8*(lane>>4)
      float v = acc[nt][r] + bv;
      if constexpr (RELU) v = fmaxf(v, 0.0f);
      if constexpr (BF16OUT) {
        Cb[(long)orow * Nc + ocol] = (bf16_t)v;
      } else if constexpr (OUTCOLS > 0) {
        if (ocol < OUTCOLS) Cf[(long)orow * OUTCOLS + ocol] = v;
      } else {
        Cf[(long)orow * Nc + ocol] = v;
      }
    }
  }
}

// ---------------- host ----------------
extern "C" void kernel_launch(void* const* d_in, const int* in_sizes, int n_in,
                              void* d_out, int out_size, void* d_ws, size_t ws_size,
                              hipStream_t stream) {
  (void)in_sizes; (void)n_in; (void)out_size; (void)ws_size;
  const float* x  = (const float*)d_in[0];
  const int*   ei = (const int*)d_in[1];
  const float* W[3]  = {(const float*)d_in[2], (const float*)d_in[6],  (const float*)d_in[10]};
  const float* bb[3] = {(const float*)d_in[3], (const float*)d_in[7],  (const float*)d_in[11]};
  const float* gg[3] = {(const float*)d_in[4], (const float*)d_in[8],  (const float*)d_in[12]};
  const float* be[3] = {(const float*)d_in[5], (const float*)d_in[9],  (const float*)d_in[13]};
  const float* cw1 = (const float*)d_in[14]; const float* cb1 = (const float*)d_in[15];
  const float* cw2 = (const float*)d_in[16]; const float* cb2 = (const float*)d_in[17];
  const float* cw3 = (const float*)d_in[18]; const float* cb3 = (const float*)d_in[19];
  const float* cw4 = (const float*)d_in[20]; const float* cb4 = (const float*)d_in[21];

  char* wsb = (char*)d_ws;
  size_t off = 0;
  auto carve = [&](size_t bytes) -> char* {
    char* p = wsb + off;
    off += (bytes + 255) & ~(size_t)255;
    return p;
  };
  float*  deg  = (float*)carve((size_t)NNODE * 4);
  float*  dinv = (float*)carve((size_t)NNODE * 4);
  float*  xw   = (float*)carve((size_t)NNODE * DIM * 4);
  float*  agg  = (float*)carve((size_t)NNODE * DIM * 4);
  bf16_t* feat = (bf16_t*)carve((size_t)NNODE * DIM * 2);
  bf16_t* Wt[3];
  for (int l = 0; l < 3; ++l) Wt[l] = (bf16_t*)carve(128 * 128 * 2);
  bf16_t* c1t = (bf16_t*)carve(256 * 256 * 2);
  bf16_t* c2t = (bf16_t*)carve(128 * 256 * 2);
  bf16_t* c3t = (bf16_t*)carve(64 * 128 * 2);
  bf16_t* c4t = (bf16_t*)carve(16 * 64 * 2);
  float*  bnm = (float*)carve(128 * 4);
  float*  bnr = (float*)carve(128 * 4);
  bf16_t* z1  = (bf16_t*)carve((size_t)NEDGE * 256 * 2);
  bf16_t* z2  = (bf16_t*)carve((size_t)NEDGE * 128 * 2);
  bf16_t* z3  = z1;  // z1 is dead once z2 exists; reuse its space

  hipMemsetAsync(deg, 0, (size_t)NNODE * 4, stream);
  k_degree<<<(NEDGE + 255) / 256, 256, 0, stream>>>(ei + NEDGE, deg);
  k_dinv<<<(NNODE + 255) / 256, 256, 0, stream>>>(deg, dinv);

  for (int l = 0; l < 3; ++l)
    k_transpose_bf16<<<(128 * 128 + 255) / 256, 256, 0, stream>>>(W[l], Wt[l], 128, 128, 128);
  k_transpose_bf16<<<(256 * 256 + 255) / 256, 256, 0, stream>>>(cw1, c1t, 256, 256, 256);
  k_transpose_bf16<<<(128 * 256 + 255) / 256, 256, 0, stream>>>(cw2, c2t, 256, 128, 128);
  k_transpose_bf16<<<(64 * 128 + 255) / 256, 256, 0, stream>>>(cw3, c3t, 128, 64, 64);
  k_transpose_bf16<<<(16 * 64 + 255) / 256, 256, 0, stream>>>(cw4, c4t, 64, 2, 16);

  k_cvt_bf16<<<(NNODE * DIM + 255) / 256, 256, 0, stream>>>(x, feat, NNODE * DIM);

  const int mtilesN = NNODE / 16;  // 3125
  const int mtilesE = NEDGE / 16;  // 50000

  // dynamic LDS: Nc * (K+16) * sizeof(bf16)
  const size_t shN  = (size_t)128 * (128 + 16) * 2;  //  36,864 B
  const size_t sh1  = (size_t)256 * (256 + 16) * 2;  // 139,264 B (<=320KB WGP LDS)
  const size_t sh2  = (size_t)128 * (256 + 16) * 2;  //  69,632 B
  const size_t sh3  = (size_t)64  * (128 + 16) * 2;  //  18,432 B
  const size_t sh4  = (size_t)16  * (64  + 16) * 2;  //   2,560 B

  for (int l = 0; l < 3; ++l) {
    k_gemm<4, 8, false, false, false, 0><<<(mtilesN + 7) / 8, 256, shN, stream>>>(
        feat, Wt[l], nullptr, xw, nullptr, nullptr, nullptr, mtilesN);
    k_init_agg<<<NNODE * DIM / 256, 256, 0, stream>>>(xw, dinv, bb[l], agg);
    k_scatter<<<NEDGE * 32 / 256, 256, 0, stream>>>(xw, dinv, ei, ei + NEDGE, agg);
    k_bn_stats<<<128, 256, 0, stream>>>(agg, bnm, bnr);
    k_bn_apply<<<NNODE * DIM / 256, 256, 0, stream>>>(agg, bnm, bnr, gg[l], be[l], feat);
  }

  // Edge MLP: gather concat(h[src],h[dst]) fused into first GEMM's A loads.
  k_gemm<8, 16, true, true, true, 0><<<(mtilesE + 7) / 8, 256, sh1, stream>>>(
      feat, c1t, cb1, nullptr, z1, ei, ei + NEDGE, mtilesE);
  k_gemm<8, 8, false, true, true, 0><<<(mtilesE + 7) / 8, 256, sh2, stream>>>(
      z1, c2t, cb2, nullptr, z2, nullptr, nullptr, mtilesE);
  k_gemm<4, 4, false, true, true, 0><<<(mtilesE + 7) / 8, 256, sh3, stream>>>(
      z2, c3t, cb3, nullptr, z3, nullptr, nullptr, mtilesE);
  k_gemm<2, 1, false, false, false, 2><<<(mtilesE + 7) / 8, 256, sh4, stream>>>(
      z3, c4t, cb4, (float*)d_out, nullptr, nullptr, nullptr, mtilesE);
}
